// Controller_76355928588609
// MI455X (gfx1250) — matrix operand order repaired and checked by reading, not compiled, not run on hardware
//
#include <hip/hip_runtime.h>

#define B_  128
#define S_  256
#define H_  1024
#define Z_  256
#define T_  205            // int(256*0.8)+1
#define KT_ (Z_ + H_)      // 1280: concat [z | h] contraction dim

typedef __attribute__((ext_vector_type(16))) __bf16 v16bf;
typedef __attribute__((ext_vector_type(8)))  __bf16 v8bf;
typedef __attribute__((ext_vector_type(8)))  float  v8f;

// ---- WMMA fragment helpers (16x16x32 bf16, wave32 layouts per CDNA5 ISA) ----

// A (16xK row-major bf16 in LDS): lane<16 -> row=lane, K-bytes {k0..k0+7, k0+16..k0+23}
//                                 lane>=16 -> row=lane-16, {k0+8..k0+15, k0+24..k0+31}
static __device__ __forceinline__ v16bf load_a_frag(const __bf16* __restrict__ Alds,
                                                    int Kt, int k0, int lane) {
  const int r  = lane & 15;
  const int kb = (lane >> 4) << 3;   // 0 or 8
  const __bf16* p = Alds + r * Kt + k0 + kb;
  v8bf lo = *(const v8bf*)(p);
  v8bf hi = *(const v8bf*)(p + 16);
  v16bf a;
#pragma unroll
  for (int i = 0; i < 8; ++i) { a[i] = lo[i]; a[i + 8] = hi[i]; }
  return a;
}

// B = W^T with W row-major [N,K] bf16: lane<16 -> col=n0+lane, K={k0..k0+7, k0+16..k0+23}
//                                      lane>=16 -> col=n0+lane-16, K={k0+8..k0+15, k0+24..k0+31}
static __device__ __forceinline__ v16bf load_b_frag(const __bf16* __restrict__ W,
                                                    int n0, int Kt, int k0, int lane) {
  const int kb = (lane >> 4) << 3;   // 0 or 8
  const __bf16* p = W + (size_t)(n0 + (lane & 15)) * Kt + k0 + kb;
  v8bf lo = *(const v8bf*)(p);
  v8bf hi = *(const v8bf*)(p + 16);
  v16bf b;
#pragma unroll
  for (int i = 0; i < 8; ++i) { b[i] = lo[i]; b[i + 8] = hi[i]; }
  return b;
}

static __device__ __forceinline__ v8f wmma_bf16(v16bf a, v16bf b, v8f c) {
  return __builtin_amdgcn_wmma_f32_16x16x32_bf16(false, a, false, b, (short)0, c,
                                                 false, false);
}

static __device__ __forceinline__ float sigmoid_f(float x) {
  return 1.0f / (1.0f + __expf(-x));
}

// ---- Kernel 1: fused gates GEMM + LSTM cell ----------------------------------------
// grid (B/16, H/64), 256 threads = 8 waves.
// Wave w: gate = w>>1, 32 h-cols starting at (w&1)*32 -> two 16-wide WMMA tiles.
__global__ __launch_bounds__(256) void k_gates_lstm(
    const float* __restrict__ zin,   // [B,Z]
    const float* __restrict__ hin,   // [B,H] (previous h)
    float* __restrict__ hout,        // [B,H] (new h; distinct buffer)
    float* __restrict__ cbuf,        // [B,H] in/out (exclusive per WG tile)
    const __bf16* __restrict__ Wg,   // [4H, KT] row-major (W_ih | W_hh) bf16
    const float* __restrict__ bias)  // [4H] = b_ih + b_hh
{
  __shared__ __bf16 Alds[16 * KT_];  // 40 KB: staged bf16 [z|h] rows
  __shared__ float  Glds[16 * 256];  // 16 KB: gate tile (4 gates x 64 cols)

  const int m0   = blockIdx.x * 16;
  const int h0   = blockIdx.y * 64;
  const int tid  = threadIdx.x;
  const int wv   = tid >> 5;
  const int lane = tid & 31;

  for (int idx = tid; idx < 16 * KT_; idx += 256) {
    const int r = idx / KT_;
    const int k = idx - r * KT_;
    const float v = (k < Z_) ? zin[(size_t)(m0 + r) * Z_ + k]
                             : hin[(size_t)(m0 + r) * H_ + (k - Z_)];
    Alds[idx] = (__bf16)v;
  }
  __syncthreads();

  const int gate = wv >> 1;
  const int colw = (wv & 1) * 32;
  const int n0   = gate * H_ + h0 + colw;    // global gate column

  v8f acc0 = {}; v8f acc1 = {};
  for (int k0 = 0; k0 < KT_; k0 += 32) {
    __builtin_prefetch(Wg + (size_t)(n0 + (lane & 15)) * KT_ + k0 + 64, 0, 1);
    v16bf a  = load_a_frag(Alds, KT_, k0, lane);
    v16bf b0 = load_b_frag(Wg, n0,      KT_, k0, lane);
    v16bf b1 = load_b_frag(Wg, n0 + 16, KT_, k0, lane);
    acc0 = wmma_bf16(a, b0, acc0);
    acc1 = wmma_bf16(a, b1, acc1);
  }

  // C layout: VGPR v holds row (lane<16 ? v : v+8), col = lane&15
  const int lc = gate * 64 + colw + (lane & 15);
  const int rb = (lane >> 4) * 8;
#pragma unroll
  for (int v = 0; v < 8; ++v) {
    Glds[(rb + v) * 256 + lc]      = acc0[v];
    Glds[(rb + v) * 256 + lc + 16] = acc1[v];
  }
  __syncthreads();

  // fused LSTM elementwise over 16x64 h tile
  for (int idx = tid; idx < 16 * 64; idx += 256) {
    const int r  = idx >> 6;
    const int hc = idx & 63;
    const int hg = h0 + hc;
    const float iv = Glds[r * 256 +       hc] + bias[         hg];
    const float fv = Glds[r * 256 +  64 + hc] + bias[    H_ + hg];
    const float gv = Glds[r * 256 + 128 + hc] + bias[2 * H_ + hg];
    const float ov = Glds[r * 256 + 192 + hc] + bias[3 * H_ + hg];
    const size_t off = (size_t)(m0 + r) * H_ + hg;
    const float cp = cbuf[off];
    const float cn = sigmoid_f(fv) * cp + sigmoid_f(iv) * tanhf(gv);
    cbuf[off] = cn;
    hout[off] = sigmoid_f(ov) * tanhf(cn);
  }
}

// ---- Kernel 2: generic C[128,N] = A[128,K] @ W[N,K]^T (bf16 WMMA, f32 out) ---------
// grid (B/16, N/128), 128 threads = 4 waves; wave w owns 32 output columns.
__global__ __launch_bounds__(128) void k_gemm_bf16(
    const float* __restrict__ A, int K,
    const __bf16* __restrict__ W,
    float* __restrict__ C, int ldc,
    float* __restrict__ C2, long long c2_stride)   // optional second destination
{
  extern __shared__ __bf16 Alds[];   // 16*K bf16
  const int m0   = blockIdx.x * 16;
  const int n0   = blockIdx.y * 128;
  const int tid  = threadIdx.x;
  const int wv   = tid >> 5;
  const int lane = tid & 31;

  for (int idx = tid; idx < 16 * K; idx += 128)
    Alds[idx] = (__bf16)A[(size_t)(m0 + idx / K) * K + (idx % K)];
  __syncthreads();

  const int nw = n0 + wv * 32;
  v8f acc0 = {}; v8f acc1 = {};
  for (int k0 = 0; k0 < K; k0 += 32) {
    __builtin_prefetch(W + (size_t)(nw + (lane & 15)) * K + k0 + 64, 0, 1);
    v16bf a  = load_a_frag(Alds, K, k0, lane);
    v16bf b0 = load_b_frag(W, nw,      K, k0, lane);
    v16bf b1 = load_b_frag(W, nw + 16, K, k0, lane);
    acc0 = wmma_bf16(a, b0, acc0);
    acc1 = wmma_bf16(a, b1, acc1);
  }

  const int col = nw + (lane & 15);
  const int rb  = (lane >> 4) * 8;
#pragma unroll
  for (int v = 0; v < 8; ++v) {
    const int row = m0 + rb + v;
    C[(size_t)row * ldc + col]      = acc0[v];
    C[(size_t)row * ldc + col + 16] = acc1[v];
    if (C2) {
      C2[(size_t)row * c2_stride + col]      = acc0[v];
      C2[(size_t)row * c2_stride + col + 16] = acc1[v];
    }
  }
}

// ---- Kernel 3: per-batch attention (scores -> softmax -> context), VALU ------------
// grid (B), 256 threads; thread s owns score s (S_ == 256).
__global__ __launch_bounds__(256) void k_attention(
    const float* __restrict__ enc,   // [B,S,H]
    const float* __restrict__ x,     // [B,H] projected decoder hidden
    float* __restrict__ cxt)         // [B,H]
{
  __shared__ float xl[H_];
  __shared__ float pl[S_];
  __shared__ float red[256];
  const int b   = blockIdx.x;
  const int tid = threadIdx.x;
  const float* eb = enc + (size_t)b * S_ * H_;

  for (int e = tid; e < H_; e += 256) xl[e] = x[(size_t)b * H_ + e];
  __syncthreads();

  float sc = 0.f;
  {
    const float* row = eb + (size_t)tid * H_;
    for (int e = 0; e < H_; e += 4) {
      const float4 v = *(const float4*)(row + e);
      sc += v.x * xl[e] + v.y * xl[e + 1] + v.z * xl[e + 2] + v.w * xl[e + 3];
    }
  }
  red[tid] = sc; __syncthreads();
  for (int s2 = 128; s2 > 0; s2 >>= 1) {
    if (tid < s2) red[tid] = fmaxf(red[tid], red[tid + s2]);
    __syncthreads();
  }
  const float mx = red[0]; __syncthreads();
  const float ex = __expf(sc - mx);
  red[tid] = ex; __syncthreads();
  for (int s2 = 128; s2 > 0; s2 >>= 1) {
    if (tid < s2) red[tid] += red[tid + s2];
    __syncthreads();
  }
  pl[tid] = ex / red[0];
  __syncthreads();

  const int e0 = tid * 4;   // 256 threads * 4 = H_
  float a0 = 0.f, a1 = 0.f, a2 = 0.f, a3 = 0.f;
  for (int s = 0; s < S_; ++s) {
    const float p = pl[s];
    const float* row = eb + (size_t)s * H_ + e0;
    a0 += p * row[0]; a1 += p * row[1]; a2 += p * row[2]; a3 += p * row[3];
  }
  float* dst = cxt + (size_t)b * H_ + e0;
  dst[0] = a0; dst[1] = a1; dst[2] = a2; dst[3] = a3;
}

// ---- Prep kernels ------------------------------------------------------------------
__global__ void k_prep_gates(const float* __restrict__ Wih, const float* __restrict__ Whh,
                             const float* __restrict__ bih, const float* __restrict__ bhh,
                             __bf16* __restrict__ Wg, float* __restrict__ bias) {
  const size_t idx = (size_t)blockIdx.x * 256 + threadIdx.x;
  if (idx < (size_t)4 * H_ * KT_) {
    const int n = (int)(idx / KT_);
    const int k = (int)(idx % KT_);
    const float v = (k < Z_) ? Wih[(size_t)n * Z_ + k] : Whh[(size_t)n * H_ + (k - Z_)];
    Wg[idx] = (__bf16)v;
  }
  if (idx < (size_t)4 * H_) bias[idx] = bih[idx] + bhh[idx];
}

__global__ void k_cvt_bf16(const float* __restrict__ src, __bf16* __restrict__ dst, size_t n) {
  const size_t idx = (size_t)blockIdx.x * 256 + threadIdx.x;
  if (idx < n) dst[idx] = (__bf16)src[idx];
}

__global__ void k_init(float* __restrict__ h0, float* __restrict__ c0,
                       float* __restrict__ z0, const float* __restrict__ init_z) {
  const size_t idx = (size_t)blockIdx.x * 256 + threadIdx.x;
  if (idx < (size_t)B_ * H_) { h0[idx] = 0.f; c0[idx] = 0.f; }
  if (idx < (size_t)B_ * Z_) z0[idx] = init_z[idx % Z_];
}

// ---- Host --------------------------------------------------------------------------
extern "C" void kernel_launch(void* const* d_in, const int* in_sizes, int n_in,
                              void* d_out, int out_size, void* d_ws, size_t ws_size,
                              hipStream_t stream) {
  (void)in_sizes; (void)n_in; (void)out_size; (void)ws_size;
  const float* enc    = (const float*)d_in[0];
  const float* W_ih   = (const float*)d_in[1];
  const float* W_hh   = (const float*)d_in[2];
  const float* b_ih   = (const float*)d_in[3];
  const float* b_hh   = (const float*)d_in[4];
  const float* W_attn = (const float*)d_in[5];
  const float* W_mu   = (const float*)d_in[6];
  const float* init_z = (const float*)d_in[8];   // d_in[7] = W_sg (dead in eval)
  float* out = (float*)d_out;

  char* ws = (char*)d_ws;
  size_t off = 0;
  auto carve = [&](size_t bytes) -> char* {
    char* p = ws + off;
    off += (bytes + 255) & ~(size_t)255;
    return p;
  };
  __bf16* Wg   = (__bf16*)carve((size_t)4 * H_ * KT_ * sizeof(__bf16)); // 10.5 MB
  float*  bias = (float*) carve((size_t)4 * H_ * sizeof(float));
  __bf16* Wab  = (__bf16*)carve((size_t)H_ * H_ * sizeof(__bf16));      // 2 MB
  __bf16* Wmb  = (__bf16*)carve((size_t)Z_ * H_ * sizeof(__bf16));      // 0.5 MB
  float*  ha   = (float*) carve((size_t)B_ * H_ * sizeof(float));
  float*  hb   = (float*) carve((size_t)B_ * H_ * sizeof(float));
  float*  cb   = (float*) carve((size_t)B_ * H_ * sizeof(float));
  float*  zb   = (float*) carve((size_t)B_ * Z_ * sizeof(float));
  float*  xb   = (float*) carve((size_t)B_ * H_ * sizeof(float));
  float*  cx   = (float*) carve((size_t)B_ * H_ * sizeof(float));

  k_prep_gates<<<(4 * H_ * KT_ + 255) / 256, 256, 0, stream>>>(W_ih, W_hh, b_ih, b_hh, Wg, bias);
  k_cvt_bf16<<<(H_ * H_ + 255) / 256, 256, 0, stream>>>(W_attn, Wab, (size_t)H_ * H_);
  k_cvt_bf16<<<(Z_ * H_ + 255) / 256, 256, 0, stream>>>(W_mu, Wmb, (size_t)Z_ * H_);
  k_init<<<(B_ * H_ + 255) / 256, 256, 0, stream>>>(ha, cb, zb, init_z);

  const size_t smem_gemm = (size_t)16 * H_ * sizeof(__bf16);  // 32 KB A-stage
  for (int t = 0; t < T_; ++t) {
    const float* hin  = (t & 1) ? hb : ha;
    float*       hout = (t & 1) ? ha : hb;
    // gates + LSTM cell (WMMA over K=1280)
    k_gates_lstm<<<dim3(B_ / 16, H_ / 64), 256, 0, stream>>>(zb, hin, hout, cb, Wg, bias);
    // x = h_new @ W_attn^T
    k_gemm_bf16<<<dim3(B_ / 16, H_ / 128), 128, smem_gemm, stream>>>(
        hout, H_, Wab, xb, H_, nullptr, 0);
    // softmax attention + context
    k_attention<<<B_, 256, 0, stream>>>(enc, xb, cx);
    // mu = cxt @ W_mu^T  ->  z (next step) and d_out[:, t, :]
    k_gemm_bf16<<<dim3(B_ / 16, Z_ / 128), 128, smem_gemm, stream>>>(
        cx, H_, Wmb, zb, Z_, out + (size_t)t * Z_, (long long)T_ * Z_);
  }
}
